// GAT_4681514353033
// MI455X (gfx1250) — compile-verified, matrix operand order checked by reading, not run on hardware
//
#include <hip/hip_runtime.h>
#include <hip/hip_bf16.h>
#include <cstddef>

// ---------------------------------------------------------------------------
// Types for CDNA5 WMMA (wave32): v_wmma_f32_16x16x32_bf16
// ---------------------------------------------------------------------------
typedef __attribute__((ext_vector_type(8)))  float   v8f;
typedef __attribute__((ext_vector_type(8)))  __bf16  v8bf;
typedef __attribute__((ext_vector_type(16))) __bf16  v16bf;

#define LRELU_SLOPE 0.2f

// Build a 16x32 bf16 A/B fragment for one lane.
// Per ISA 16-bit A layout: lane m = lane%16, half = lane/16;
// VGPR0..3 hold K = half*8 .. half*8+7 (contiguous), VGPR4..7 hold K+16..K+23.
// => two contiguous 16-byte LDS loads per lane.
__device__ __forceinline__ v16bf load_frag16(const __bf16* p) {
  v8bf lo = *(const v8bf*)(p);
  v8bf hi = *(const v8bf*)(p + 16);
  v16bf r;
#pragma unroll
  for (int i = 0; i < 8; ++i) { r[i] = lo[i]; r[i + 8] = hi[i]; }
  return r;
}

// Ordered-uint encoding so unsigned atomicMax == float max (handles negatives).
__device__ __forceinline__ unsigned fenc(float f) {
  unsigned u = __float_as_uint(f);
  return (u >> 31) ? ~u : (u | 0x80000000u);
}
__device__ __forceinline__ float fdec(unsigned u) {
  return __uint_as_float((u >> 31) ? (u ^ 0x80000000u) : ~u);
}
#define ENC_NEG_INF 0x007FFFFFu   // fenc(-inf)

// ---------------------------------------------------------------------------
// GEMM: out[N,64] = A[N,K] @ W[K,64], fp32 in/out, bf16x3 WMMA inside.
// Block: 128 threads (4 waves); each block does 64 rows; each wave a 16-row
// strip across all 64 output columns (4 wmma tiles), K streamed 32 at a time.
// ---------------------------------------------------------------------------
__global__ __launch_bounds__(128)
void gat_gemm64_wmma(const float* __restrict__ A, const float* __restrict__ W,
                     float* __restrict__ out, int nRows, int K) {
  __shared__ __bf16 At_hi[64 * 32];
  __shared__ __bf16 At_lo[64 * 32];
  __shared__ __bf16 Bt_hi[64 * 32];   // transposed: Bt[n*32 + k]
  __shared__ __bf16 Bt_lo[64 * 32];

  const int tid  = threadIdx.x;
  const int lane = tid & 31;
  const int wv   = tid >> 5;          // wave 0..3
  const int row0 = blockIdx.x * 64;
  const int m0   = wv * 16;
  const int lm   = lane & 15;
  const int half = lane >> 4;
  const int offA = half * 8;          // element offset of first contiguous run

  v8f acc[4];
#pragma unroll
  for (int j = 0; j < 4; ++j)
#pragma unroll
    for (int r = 0; r < 8; ++r) acc[j][r] = 0.0f;

  for (int kk = 0; kk < K; kk += 32) {
    __syncthreads();
    // ---- stage A tile (64x32) and B panel (32x64, transposed) to LDS ------
#pragma unroll
    for (int it = 0; it < 4; ++it) {
      int c = tid + it * 128;               // 0..511 float4 chunks
      // A: row r, cols c4..c4+3
      int r  = c >> 3;
      int c4 = (c & 7) << 2;
      int gr = row0 + r;
      float4 va = make_float4(0.f, 0.f, 0.f, 0.f);
      if (gr < nRows) {
        va = *(const float4*)(A + (size_t)gr * K + kk + c4);
        __builtin_prefetch(A + (size_t)gr * K + kk + 32, 0, 0);
      }
      {
        __bf16* dh = &At_hi[r * 32 + c4];
        __bf16* dl = &At_lo[r * 32 + c4];
        float f[4] = {va.x, va.y, va.z, va.w};
#pragma unroll
        for (int i = 0; i < 4; ++i) {
          __bf16 h = (__bf16)f[i];
          dh[i] = h;
          dl[i] = (__bf16)(f[i] - (float)h);
        }
      }
      // B: k row, 4 columns (transpose into Bt[n*32 + k])
      int k  = c >> 4;
      int n4 = (c & 15) << 2;
      float4 vb = *(const float4*)(W + (size_t)(kk + k) * 64 + n4);
      float fb[4] = {vb.x, vb.y, vb.z, vb.w};
#pragma unroll
      for (int i = 0; i < 4; ++i) {
        __bf16 h = (__bf16)fb[i];
        Bt_hi[(n4 + i) * 32 + k] = h;
        Bt_lo[(n4 + i) * 32 + k] = (__bf16)(fb[i] - (float)h);
      }
    }
    __syncthreads();

    // ---- WMMA: acc += Ahi*Bhi + Ahi*Blo + Alo*Bhi (bf16x3, ~fp32 accurate)
    v16bf ah = load_frag16(&At_hi[(m0 + lm) * 32 + offA]);
    v16bf al = load_frag16(&At_lo[(m0 + lm) * 32 + offA]);
#pragma unroll
    for (int j = 0; j < 4; ++j) {
      v16bf bh = load_frag16(&Bt_hi[(j * 16 + lm) * 32 + offA]);
      v16bf bl = load_frag16(&Bt_lo[(j * 16 + lm) * 32 + offA]);
      acc[j] = __builtin_amdgcn_wmma_f32_16x16x32_bf16(
                   false, ah, false, bh, (short)0, acc[j], false, false);
      acc[j] = __builtin_amdgcn_wmma_f32_16x16x32_bf16(
                   false, ah, false, bl, (short)0, acc[j], false, false);
      acc[j] = __builtin_amdgcn_wmma_f32_16x16x32_bf16(
                   false, al, false, bh, (short)0, acc[j], false, false);
    }
  }

  // ---- epilogue: C/D layout — VGPR r: lanes0-15 M=r, lanes16-31 M=r+8 -----
#pragma unroll
  for (int r = 0; r < 8; ++r) {
    int gm = row0 + m0 + r + half * 8;
    if (gm < nRows) {
#pragma unroll
      for (int j = 0; j < 4; ++j)
        out[(size_t)gm * 64 + j * 16 + lm] = acc[j][r];
    }
  }
}

// ---------------------------------------------------------------------------
// Utility / elementwise kernels
// ---------------------------------------------------------------------------
__global__ void gat_fill_u32(unsigned* __restrict__ p, unsigned v, long n) {
  long i = (long)blockIdx.x * blockDim.x + threadIdx.x;
  if (i < n) p[i] = v;
}

// per-node half scores: sc[n,h] = sum_c h[n, h*C+c] * a[h*C+c]
__global__ void gat_scores(const float* __restrict__ h,
                           const float* __restrict__ a_s,
                           const float* __restrict__ a_d,
                           float* __restrict__ sc_s, float* __restrict__ sc_d,
                           int nN, int H, int C) {
  int t = blockIdx.x * blockDim.x + threadIdx.x;
  if (t >= nN * H) return;
  int n = t / H, hh = t - n * H;
  const float* row = h + (size_t)n * 64 + hh * C;
  const float* as = a_s + hh * C;
  const float* ad = a_d + hh * C;
  float ss = 0.f, sd = 0.f;
  for (int c = 0; c < C; ++c) { ss += row[c] * as[c]; sd += row[c] * ad[c]; }
  sc_s[t] = ss; sc_d[t] = sd;
}

__device__ __forceinline__ void edge_nodes(const int* ei, int E, int e,
                                           int& s, int& d) {
  if (e < E) { s = ei[e]; d = ei[E + e]; }
  else       { s = d = e - E; }               // appended self-loops
}

__device__ __forceinline__ float edge_logit(const float* ss, const float* sd,
                                            int s, int d, int H, int hh) {
  float lg = ss[s * H + hh] + sd[d * H + hh];
  return (lg >= 0.f) ? lg : LRELU_SLOPE * lg;
}

__global__ void gat_edge_max(const int* __restrict__ ei, int E, int Etot,
                             const float* __restrict__ ss,
                             const float* __restrict__ sd,
                             unsigned* __restrict__ mx, int H) {
  int t = blockIdx.x * blockDim.x + threadIdx.x;
  if (t >= Etot * H) return;
  int e = t / H, hh = t - e * H;
  int s, d; edge_nodes(ei, E, e, s, d);
  atomicMax(&mx[d * H + hh], fenc(edge_logit(ss, sd, s, d, H, hh)));
}

__global__ void gat_edge_sum(const int* __restrict__ ei, int E, int Etot,
                             const float* __restrict__ ss,
                             const float* __restrict__ sd,
                             const unsigned* __restrict__ mx,
                             float* __restrict__ denom, int H) {
  int t = blockIdx.x * blockDim.x + threadIdx.x;
  if (t >= Etot * H) return;
  int e = t / H, hh = t - e * H;
  int s, d; edge_nodes(ei, E, e, s, d);
  float lg = edge_logit(ss, sd, s, d, H, hh);
  atomicAdd(&denom[d * H + hh], expf(lg - fdec(mx[d * H + hh])));
}

// 8 threads per edge; thread q handles channels [q*8, q*8+8).
// For H=8 head(q)=q; for H=1 head=0.
__global__ void gat_edge_agg(const int* __restrict__ ei, int E, int Etot,
                             const float* __restrict__ ss,
                             const float* __restrict__ sd,
                             const unsigned* __restrict__ mx,
                             const float* __restrict__ denom,
                             const float* __restrict__ h,
                             float* __restrict__ agg, int H) {
  long t = (long)blockIdx.x * blockDim.x + threadIdx.x;
  if (t >= (long)Etot * 8) return;
  int e = (int)(t >> 3), q = (int)(t & 7);
  int hh = (q * H) >> 3;
  int s, d; edge_nodes(ei, E, e, s, d);
  float lg = edge_logit(ss, sd, s, d, H, hh);
  float ex = expf(lg - fdec(mx[d * H + hh]));
  float alpha = ex / (denom[d * H + hh] + 1e-16f);
  const float* src = h + (size_t)s * 64 + q * 8;
  float* dst = agg + (size_t)d * 64 + q * 8;
#pragma unroll
  for (int c = 0; c < 8; ++c) atomicAdd(&dst[c], src[c] * alpha);
}

// x = elu(x + b[col])  (in place), n = N*64
__global__ void gat_elu_bias(float* __restrict__ x, const float* __restrict__ b,
                             long n) {
  long i = (long)blockIdx.x * blockDim.x + threadIdx.x;
  if (i >= n) return;
  float v = x[i] + b[(int)(i & 63)];
  x[i] = (v > 0.f) ? v : (expf(v) - 1.f);
}

// out[n,:] = log_softmax(out[n,:] + b), 64 classes, in place
__global__ void gat_logsoftmax64(float* __restrict__ o,
                                 const float* __restrict__ b, int nN) {
  int n = blockIdx.x * blockDim.x + threadIdx.x;
  if (n >= nN) return;
  float r[64];
  float* row = o + (size_t)n * 64;
#pragma unroll
  for (int c = 0; c < 64; ++c) r[c] = row[c] + b[c];
  float m = r[0];
#pragma unroll
  for (int c = 1; c < 64; ++c) m = fmaxf(m, r[c]);
  float s = 0.f;
#pragma unroll
  for (int c = 0; c < 64; ++c) s += expf(r[c] - m);
  float ls = m + logf(s);
#pragma unroll
  for (int c = 0; c < 64; ++c) row[c] = r[c] - ls;
}

// ---------------------------------------------------------------------------
// Launch
// ---------------------------------------------------------------------------
static inline int cdiv(long a, int b) { return (int)((a + b - 1) / b); }

extern "C" void kernel_launch(void* const* d_in, const int* in_sizes, int n_in,
                              void* d_out, int out_size, void* d_ws, size_t ws_size,
                              hipStream_t stream) {
  const float* x     = (const float*)d_in[0];
  const int*   ei    = (const int*)d_in[1];
  const float* W1    = (const float*)d_in[2];
  const float* a_s1  = (const float*)d_in[3];
  const float* a_d1  = (const float*)d_in[4];
  const float* b1    = (const float*)d_in[5];
  const float* W2    = (const float*)d_in[6];
  const float* a_s2  = (const float*)d_in[7];
  const float* a_d2  = (const float*)d_in[8];
  const float* b2    = (const float*)d_in[9];
  float* out = (float*)d_out;

  const int N    = in_sizes[0] / 512;
  const int E    = in_sizes[1] / 2;
  const int Etot = E + N;

  // workspace layout (floats)
  float*    h1    = (float*)d_ws;                  // [N,64]  h1 -> later h2
  float*    agg   = h1 + (size_t)N * 64;           // [N,64]  agg1 -> x2 (in place)
  float*    sc_s  = agg + (size_t)N * 64;          // [N,8]   (layer2 uses [N,1])
  float*    sc_d  = sc_s + (size_t)N * 8;          // [N,8]
  unsigned* mx    = (unsigned*)(sc_d + (size_t)N * 8); // [N,8] encoded
  float*    denom = (float*)(mx + (size_t)N * 8);  // [N,8]

  const int TB = 256;

  // ===== layer 1: 8 heads x 8 ch =====
  gat_gemm64_wmma<<<dim3(cdiv(N, 64)), dim3(128), 0, stream>>>(x, W1, h1, N, 512);
  gat_scores<<<cdiv((long)N * 8, TB), TB, 0, stream>>>(h1, a_s1, a_d1, sc_s, sc_d, N, 8, 8);

  gat_fill_u32<<<cdiv((long)N * 8, TB), TB, 0, stream>>>(mx, ENC_NEG_INF, (long)N * 8);
  gat_fill_u32<<<cdiv((long)N * 8, TB), TB, 0, stream>>>((unsigned*)denom, 0u, (long)N * 8);
  gat_fill_u32<<<cdiv((long)N * 64, TB), TB, 0, stream>>>((unsigned*)agg, 0u, (long)N * 64);

  gat_edge_max<<<cdiv((long)Etot * 8, TB), TB, 0, stream>>>(ei, E, Etot, sc_s, sc_d, mx, 8);
  gat_edge_sum<<<cdiv((long)Etot * 8, TB), TB, 0, stream>>>(ei, E, Etot, sc_s, sc_d, mx, denom, 8);
  gat_edge_agg<<<cdiv((long)Etot * 8, TB), TB, 0, stream>>>(ei, E, Etot, sc_s, sc_d, mx, denom, h1, agg, 8);

  gat_elu_bias<<<cdiv((long)N * 64, TB), TB, 0, stream>>>(agg, b1, (long)N * 64);

  // ===== layer 2: 1 head x 64 ch =====
  gat_gemm64_wmma<<<dim3(cdiv(N, 64)), dim3(128), 0, stream>>>(agg, W2, h1, N, 64); // h1 := h2
  gat_scores<<<cdiv((long)N, TB), TB, 0, stream>>>(h1, a_s2, a_d2, sc_s, sc_d, N, 1, 64);

  gat_fill_u32<<<cdiv((long)N, TB), TB, 0, stream>>>(mx, ENC_NEG_INF, (long)N);
  gat_fill_u32<<<cdiv((long)N, TB), TB, 0, stream>>>((unsigned*)denom, 0u, (long)N);
  gat_fill_u32<<<cdiv((long)N * 64, TB), TB, 0, stream>>>((unsigned*)out, 0u, (long)N * 64);

  gat_edge_max<<<cdiv((long)Etot, TB), TB, 0, stream>>>(ei, E, Etot, sc_s, sc_d, mx, 1);
  gat_edge_sum<<<cdiv((long)Etot, TB), TB, 0, stream>>>(ei, E, Etot, sc_s, sc_d, mx, denom, 1);
  gat_edge_agg<<<cdiv((long)Etot * 8, TB), TB, 0, stream>>>(ei, E, Etot, sc_s, sc_d, mx, denom, h1, out, 1);

  gat_logsoftmax64<<<cdiv((long)N, TB), TB, 0, stream>>>(out, b2, N);
}
